// FullMoeRelativeAttentionCore_42958262895121
// MI455X (gfx1250) — matrix operand order repaired and checked by reading, not compiled, not run on hardware
//
#include <hip/hip_runtime.h>
#include <hip/hip_bf16.h>
#include <math.h>
#include <stdint.h>

// ---------------------------------------------------------------------------
// Problem constants (from the reference)
// ---------------------------------------------------------------------------
#define HH   16
#define EE   8
#define PP   64
#define PVV  64
#define DD   1024
#define BB   8
#define TT   512
#define RR   1023          // 2*T - 1
#define BT   4096          // B*T

typedef __bf16 bf16;
typedef __attribute__((ext_vector_type(16))) __bf16 v16bf;
typedef __attribute__((ext_vector_type(8)))  __bf16 v8bf;
typedef __attribute__((ext_vector_type(8)))  float  v8f;

static __device__ inline v8f v8f_zero() {
  v8f z = {0.f, 0.f, 0.f, 0.f, 0.f, 0.f, 0.f, 0.f};
  return z;
}

// D = A(16x32 bf16) * B(32x16 bf16) + C(16x16 f32)
static __device__ inline v8f wmma_bf16(v16bf a, v16bf b, v8f c) {
  return __builtin_amdgcn_wmma_f32_16x16x32_bf16(
      /*neg_a=*/false, a, /*neg_b=*/false, b,
      /*c_mod=*/(short)0, c, /*reuse_a=*/false, /*reuse_b=*/false);
}

// Async 16-byte global -> LDS copy (CDNA5 async path, tracked by ASYNCcnt).
// the LDS address comes from a VGPR; generic LDS pointers keep the LDS byte
// offset in their low 32 bits, so we pass that directly.
static __device__ inline void async_copy16(unsigned lds_byte_off, const void* gaddr) {
  asm volatile("global_load_async_to_lds_b128 %0, %1, off"
               :: "v"(lds_byte_off), "v"(gaddr)
               : "memory");
}
static __device__ inline void wait_async0() {
  asm volatile("s_wait_asynccnt 0" ::: "memory");
}

// A-fragment (16x32, bf16) from a row-major matrix: row stride = ld elements.
// Lane L (L<16): row M=L, K = 0..15; lane L+16: row M=L, K = 16..31.
static __device__ inline v16bf load_a_rm(const bf16* base, int ld, int lane) {
  const bf16* p = base + (size_t)(lane & 15) * ld + ((lane >> 4) << 4);
  union { v16bf v; v8bf h[2]; } u;
  u.h[0] = *(const v8bf*)p;
  u.h[1] = *(const v8bf*)(p + 8);
  return u.v;
}

// B-fragment (32x16, bf16) from "K-contiguous" storage: element (k,n) at
// base[n*ldk + k].  Lane L<16: N=L, K=0..15; lane L+16: same N, K=16..31.
static __device__ inline v16bf load_b_ck(const bf16* base, int ldk, int lane) {
  const bf16* p = base + (size_t)(lane & 15) * ldk + ((lane >> 4) << 4);
  union { v16bf v; v8bf h[2]; } u;
  u.h[0] = *(const v8bf*)p;
  u.h[1] = *(const v8bf*)(p + 8);
  return u.v;
}

// B-fragment from a pre-packed tile: tile[lane*16 + i] is lane's element i.
static __device__ inline v16bf load_b_packed(const bf16* tile, int lane) {
  const bf16* p = tile + lane * 16;
  union { v16bf v; v8bf h[2]; } u;
  u.h[0] = *(const v8bf*)p;
  u.h[1] = *(const v8bf*)(p + 8);
  return u.v;
}

// ---------------------------------------------------------------------------
// 1) f32 -> bf16 convert (with zero padding up to ntot)
// ---------------------------------------------------------------------------
__global__ void cvt_kernel(const float* __restrict__ src, bf16* __restrict__ dst,
                           int n, int ntot) {
  int i = blockIdx.x * blockDim.x + threadIdx.x;
  if (i < n)          dst[i] = (bf16)src[i];
  else if (i < ntot)  dst[i] = (bf16)0.f;
}

// ---------------------------------------------------------------------------
// 2) Pack a batch of KxN f32 matrices into WMMA-B-fragment tile order (bf16).
//    Packed: [mat][kb][nb][lane][i], K = kb*32 + (lane>>4)*16 + i,
//                                  N = nb*16 + (lane&15).
//    Source element (k,n) at src[mat*matElems + k*sk + n*sn].
// ---------------------------------------------------------------------------
__global__ void pack_b_kernel(const float* __restrict__ src, bf16* __restrict__ dst,
                              int N, int sk, int sn, int matElems, int total) {
  int idx = blockIdx.x * blockDim.x + threadIdx.x;
  if (idx >= total) return;
  int mat  = idx / matElems;
  int pidx = idx - mat * matElems;
  int i      = pidx & 15;
  int lane   = (pidx >> 4) & 31;
  int tileId = pidx >> 9;
  int nbCount = N >> 4;
  int nb = tileId % nbCount;
  int kb = tileId / nbCount;
  int k = kb * 32 + ((lane >> 4) << 4) + i;
  int n = nb * 16 + (lane & 15);
  dst[(size_t)mat * matElems + pidx] =
      (bf16)src[(size_t)mat * matElems + (size_t)k * sk + (size_t)n * sn];
}

// ---------------------------------------------------------------------------
// 3) Gating: logits = x . w_sel, sigmoid, top-2 mask.  One thread per (bt,h).
// ---------------------------------------------------------------------------
static __device__ inline void sig_top2_store(const float* acc, float* outp) {
  float g[EE];
  #pragma unroll
  for (int e = 0; e < EE; ++e) g[e] = 1.f / (1.f + __expf(-acc[e]));
  int i1 = 0;
  #pragma unroll
  for (int e = 1; e < EE; ++e) if (g[e] > g[i1]) i1 = e;
  int i2 = (i1 == 0) ? 1 : 0;
  #pragma unroll
  for (int e = 0; e < EE; ++e) {
    if (e == i1 || e == i2) continue;
    if (g[e] > g[i2]) i2 = e;
  }
  #pragma unroll
  for (int e = 0; e < EE; ++e) outp[e] = (e == i1 || e == i2) ? g[e] : 0.f;
}

__global__ void gating_kernel(const float* __restrict__ x,
                              const float* __restrict__ wso,
                              const float* __restrict__ wsv,
                              float* __restrict__ go, float* __restrict__ gv) {
  int tid = blockIdx.x * blockDim.x + threadIdx.x;   // bt*H + h
  if (tid >= BT * HH) return;
  int h  = tid & (HH - 1);
  int bt = tid >> 4;
  const float* xr = x + (size_t)bt * DD;
  float ao[EE], av[EE];
  #pragma unroll
  for (int e = 0; e < EE; ++e) { ao[e] = 0.f; av[e] = 0.f; }
  for (int d = 0; d < DD; ++d) {
    float xv = xr[d];
    #pragma unroll
    for (int e = 0; e < EE; ++e) {
      ao[e] = fmaf(xv, wso[((size_t)h * EE + e) * DD + d], ao[e]);
      av[e] = fmaf(xv, wsv[((size_t)h * EE + e) * DD + d], av[e]);
    }
  }
  sig_top2_store(ao, go + ((size_t)bt * HH + h) * EE);
  sig_top2_store(av, gv + ((size_t)bt * HH + h) * EE);
}

// ---------------------------------------------------------------------------
// 4) MoE projection: out[bt,h,p] = sum_e g[bt,h,e] * (x[bt,:] @ W[h,e,:,p]).
//    One wave per (16-token tile, head).  x tile staged into LDS with the
//    CDNA5 async global->LDS path (no VGPR round-trip), gated by ASYNCcnt.
//    vtmode==0: write bf16 [bt][h][p] ; vtmode==1: write bf16 [b][h][pv][t].
// ---------------------------------------------------------------------------
__global__ __launch_bounds__(32)
void moe_proj_kernel(const bf16* __restrict__ xbf, const bf16* __restrict__ wpack,
                     const float* __restrict__ gates, bf16* __restrict__ outp,
                     int vtmode) {
  __shared__ __align__(16) bf16 xs[16 * DD];
  int lane = threadIdx.x;
  int bt0  = blockIdx.x * 16;
  int h    = blockIdx.y;

  { // async-stage 16 x-rows (row-major, contiguous) into LDS
    const bf16* gsrc = xbf + (size_t)bt0 * DD;
    for (int i = lane; i < (16 * DD) / 8; i += 32)
      async_copy16((unsigned)(uintptr_t)(xs + (size_t)i * 8), gsrc + (size_t)i * 8);
    wait_async0();
  }
  __syncthreads();

  v8f acc[4];
  #pragma unroll
  for (int nb = 0; nb < 4; ++nb) acc[nb] = v8f_zero();

  const size_t matElems = (size_t)DD * PP;
  for (int e = 0; e < EE; ++e) {
    float gl = gates[((size_t)(bt0 + (lane & 15)) * HH + h) * EE + e];
    if (!__any(gl != 0.f)) continue;          // uniform expert skip (top-2 sparsity)

    v8f t[4];
    #pragma unroll
    for (int nb = 0; nb < 4; ++nb) t[nb] = v8f_zero();
    const bf16* wmat = wpack + ((size_t)h * EE + e) * matElems;

    for (int kb = 0; kb < DD / 32; ++kb) {
      v16bf a = load_a_rm(xs + kb * 32, DD, lane);
      #pragma unroll
      for (int nb = 0; nb < 4; ++nb) {
        v16bf b = load_b_packed(wmat + (size_t)(kb * 4 + nb) * (32 * 16), lane);
        t[nb] = wmma_bf16(a, b, t[nb]);
      }
    }
    float gf[8];
    #pragma unroll
    for (int j = 0; j < 8; ++j)
      gf[j] = gates[((size_t)(bt0 + j + 8 * (lane >> 4)) * HH + h) * EE + e];
    #pragma unroll
    for (int nb = 0; nb < 4; ++nb)
      #pragma unroll
      for (int j = 0; j < 8; ++j)
        acc[nb][j] += gf[j] * t[nb][j];
  }

  if (vtmode == 0) {
    #pragma unroll
    for (int nb = 0; nb < 4; ++nb)
      #pragma unroll
      for (int j = 0; j < 8; ++j) {
        int M = j + 8 * (lane >> 4);
        outp[((size_t)(bt0 + M) * HH + h) * PP + nb * 16 + (lane & 15)] =
            (bf16)acc[nb][j];
      }
  } else {
    int b = bt0 / TT, tl = bt0 % TT;
    #pragma unroll
    for (int nb = 0; nb < 4; ++nb)
      #pragma unroll
      for (int j = 0; j < 8; ++j) {
        int M  = j + 8 * (lane >> 4);
        int pv = nb * 16 + (lane & 15);
        outp[(((size_t)b * HH + h) * PVV + pv) * TT + tl + M] = (bf16)acc[nb][j];
      }
  }
}

// ---------------------------------------------------------------------------
// 5) k_pos = pe @ W_pos^T : (1024 x 1024 x 1024) GEMM.  W_pos is (H*P, D)
//    row-major, which is already K-contiguous for the B-fragment.
//    Output bf16 [r][h*P+p].
// ---------------------------------------------------------------------------
__global__ __launch_bounds__(32)
void kpos_kernel(const bf16* __restrict__ pebf, const bf16* __restrict__ wposbf,
                 bf16* __restrict__ kpos) {
  int lane = threadIdx.x;
  int r0 = blockIdx.x * 16;
  int n0 = blockIdx.y * 16;
  v8f acc = v8f_zero();
  for (int kb = 0; kb < DD / 32; ++kb) {
    v16bf a = load_a_rm(pebf + (size_t)r0 * DD + kb * 32, DD, lane);
    v16bf b = load_b_ck(wposbf + (size_t)n0 * DD + kb * 32, DD, lane);
    acc = wmma_bf16(a, b, acc);
  }
  #pragma unroll
  for (int j = 0; j < 8; ++j) {
    int r = r0 + j + 8 * (lane >> 4);
    kpos[(size_t)r * (HH * PP) + n0 + (lane & 15)] = (bf16)acc[j];
  }
}

// ---------------------------------------------------------------------------
// 6) Attention: flash-style online softmax per (b, h, 16-row q-tile).
//    Relative term computed as a 16x48 WMMA band E[t, r-rbase] = q . k_pos,
//    bounced through LDS and gathered on the diagonal.
// ---------------------------------------------------------------------------
__global__ __launch_bounds__(32)
void attn_kernel(const bf16* __restrict__ qbf, const bf16* __restrict__ kbf,
                 const bf16* __restrict__ vT,  const bf16* __restrict__ kpos,
                 float* __restrict__ ctx) {
  __shared__ float Es[16][48];
  __shared__ __align__(16) bf16 Ps[16][32];
  int lane = threadIdx.x;
  int t0 = blockIdx.x * 16;
  int h  = blockIdx.y;
  int b  = blockIdx.z;

  v16bf aq[2];
  #pragma unroll
  for (int kk = 0; kk < 2; ++kk)
    aq[kk] = load_a_rm(qbf + ((size_t)(b * TT + t0) * HH + h) * PP + kk * 32,
                       HH * PP, lane);

  v8f o[4];
  #pragma unroll
  for (int nb = 0; nb < 4; ++nb) o[nb] = v8f_zero();
  float m[8], l[8];
  #pragma unroll
  for (int j = 0; j < 8; ++j) { m[j] = -1e30f; l[j] = 0.f; }

  for (int sb = 0; sb <= t0 + 15; sb += 32) {
    int rbase = sb - t0 + (TT - 1) - 15;               // in [0, 496]
    v8f s0 = v8f_zero(), s1 = v8f_zero();
    v8f e0 = v8f_zero(), e1 = v8f_zero(), e2 = v8f_zero();

    #pragma unroll
    for (int kk = 0; kk < 2; ++kk) {
      v16bf bk0 = load_b_ck(kbf + ((size_t)(b * TT + sb) * HH + h) * PP + kk * 32,
                            HH * PP, lane);
      v16bf bk1 = load_b_ck(kbf + ((size_t)(b * TT + sb + 16) * HH + h) * PP + kk * 32,
                            HH * PP, lane);
      s0 = wmma_bf16(aq[kk], bk0, s0);
      s1 = wmma_bf16(aq[kk], bk1, s1);
      v16bf br0 = load_b_ck(kpos + (size_t)(rbase)      * (HH * PP) + h * PP + kk * 32,
                            HH * PP, lane);
      v16bf br1 = load_b_ck(kpos + (size_t)(rbase + 16) * (HH * PP) + h * PP + kk * 32,
                            HH * PP, lane);
      v16bf br2 = load_b_ck(kpos + (size_t)(rbase + 32) * (HH * PP) + h * PP + kk * 32,
                            HH * PP, lane);
      e0 = wmma_bf16(aq[kk], br0, e0);
      e1 = wmma_bf16(aq[kk], br1, e1);
      e2 = wmma_bf16(aq[kk], br2, e2);
    }

    __syncthreads();
    #pragma unroll
    for (int j = 0; j < 8; ++j) {
      int M = j + 8 * (lane >> 4);
      Es[M][(lane & 15)]      = e0[j];
      Es[M][16 + (lane & 15)] = e1[j];
      Es[M][32 + (lane & 15)] = e2[j];
    }
    __syncthreads();

    float p0[8], p1[8], mt[8];
    #pragma unroll
    for (int j = 0; j < 8; ++j) {
      int M  = j + 8 * (lane >> 4);
      int tg = t0 + M;
      int c0 = (lane & 15), c1 = 16 + (lane & 15);
      float sc0 = (s0[j] + Es[M][c0 - M + 15]) * 0.125f;  // 1/sqrt(P)
      float sc1 = (s1[j] + Es[M][c1 - M + 15]) * 0.125f;
      sc0 = ((sb + c0) <= tg) ? sc0 : -1e9f;
      sc1 = ((sb + c1) <= tg) ? sc1 : -1e9f;
      p0[j] = sc0; p1[j] = sc1;
      mt[j] = fmaxf(sc0, sc1);
    }
    #pragma unroll
    for (int off = 1; off < 16; off <<= 1)
      #pragma unroll
      for (int j = 0; j < 8; ++j)
        mt[j] = fmaxf(mt[j], __shfl_xor(mt[j], off, 32));

    float rs[8];
    #pragma unroll
    for (int j = 0; j < 8; ++j) {
      float mn   = fmaxf(m[j], mt[j]);
      float corr = __expf(m[j] - mn);
      l[j] *= corr;
      #pragma unroll
      for (int nb = 0; nb < 4; ++nb) o[nb][j] *= corr;
      m[j]  = mn;
      p0[j] = __expf(p0[j] - mn);
      p1[j] = __expf(p1[j] - mn);
      rs[j] = p0[j] + p1[j];
    }
    #pragma unroll
    for (int off = 1; off < 16; off <<= 1)
      #pragma unroll
      for (int j = 0; j < 8; ++j)
        rs[j] += __shfl_xor(rs[j], off, 32);
    #pragma unroll
    for (int j = 0; j < 8; ++j) l[j] += rs[j];

    __syncthreads();
    #pragma unroll
    for (int j = 0; j < 8; ++j) {
      int M = j + 8 * (lane >> 4);
      Ps[M][lane & 15]        = (bf16)p0[j];
      Ps[M][16 + (lane & 15)] = (bf16)p1[j];
    }
    __syncthreads();

    v16bf ap = load_a_rm(&Ps[0][0], 32, lane);
    #pragma unroll
    for (int nb = 0; nb < 4; ++nb) {
      v16bf bv = load_b_ck(vT + (((size_t)b * HH + h) * PVV + nb * 16) * TT + sb,
                           TT, lane);
      o[nb] = wmma_bf16(ap, bv, o[nb]);
    }
  }

  #pragma unroll
  for (int j = 0; j < 8; ++j) {
    float inv = 1.f / l[j];
    int M = j + 8 * (lane >> 4);
    #pragma unroll
    for (int nb = 0; nb < 4; ++nb)
      ctx[((size_t)(b * TT + t0 + M) * HH + h) * PVV + nb * 16 + (lane & 15)] =
          o[nb][j] * inv;
  }
}

// ---------------------------------------------------------------------------
// 7) Output MoE projection: out[bt,d] += g_qo[bt,h,e] * ctx[bt,h,:] @ W_o[h,e].
//    One wave per (16-token tile, 64-wide d-tile).  ctx tile async-staged.
// ---------------------------------------------------------------------------
__global__ __launch_bounds__(32)
void out_kernel(const float* __restrict__ ctx, const float* __restrict__ go,
                const bf16* __restrict__ wopack, float* __restrict__ outp) {
  __shared__ __align__(16) float cs[16][PVV];
  int lane = threadIdx.x;
  int bt0 = blockIdx.x * 16;
  int d0  = blockIdx.y * 64;

  v8f acc[4];
  #pragma unroll
  for (int nb = 0; nb < 4; ++nb) acc[nb] = v8f_zero();

  for (int h = 0; h < HH; ++h) {
    __syncthreads();
    // async-stage 16 ctx rows (256B each) into LDS: 256 x 16B chunks
    for (int i = lane; i < 16 * (PVV / 4); i += 32)
      async_copy16((unsigned)(uintptr_t)&cs[i >> 4][(i & 15) * 4],
                   ctx + ((size_t)(bt0 + (i >> 4)) * HH + h) * PVV + (i & 15) * 4);
    wait_async0();
    __syncthreads();

    for (int e = 0; e < EE; ++e) {
      float gl = go[((size_t)(bt0 + (lane & 15)) * HH + h) * EE + e];
      if (!__any(gl != 0.f)) continue;
      const bf16* wmat = wopack + ((size_t)h * EE + e) * ((size_t)PVV * DD);
      int row = lane & 15;
      #pragma unroll
      for (int kk = 0; kk < 2; ++kk) {
        union { v16bf v; } u;
        #pragma unroll
        for (int t2 = 0; t2 < 16; ++t2) {
          int Kk = kk * 32 + ((lane >> 4) << 4) + t2;
          u.v[t2] = (bf16)(cs[row][Kk] * gl);
        }
        #pragma unroll
        for (int nb = 0; nb < 4; ++nb) {
          int nbg = (d0 >> 4) + nb;
          v16bf bw = load_b_packed(wmat + (size_t)(kk * (DD >> 4) + nbg) * (32 * 16),
                                   lane);
          acc[nb] = wmma_bf16(u.v, bw, acc[nb]);
        }
      }
    }
  }
  #pragma unroll
  for (int nb = 0; nb < 4; ++nb)
    #pragma unroll
    for (int j = 0; j < 8; ++j) {
      int M = j + 8 * (lane >> 4);
      outp[(size_t)(bt0 + M) * DD + d0 + nb * 16 + (lane & 15)] = acc[nb][j];
    }
}

// ---------------------------------------------------------------------------
// Host launcher
// ---------------------------------------------------------------------------
extern "C" void kernel_launch(void* const* d_in, const int* in_sizes, int n_in,
                              void* d_out, int out_size, void* d_ws, size_t ws_size,
                              hipStream_t stream) {
  (void)in_sizes; (void)n_in; (void)out_size; (void)ws_size;
  const float* x    = (const float*)d_in[0];
  const float* pe   = (const float*)d_in[1];
  const float* wso  = (const float*)d_in[2];
  const float* wsv  = (const float*)d_in[3];
  const float* Wq   = (const float*)d_in[4];
  const float* Wk   = (const float*)d_in[5];
  const float* Wv   = (const float*)d_in[6];
  const float* Wo   = (const float*)d_in[7];
  const float* Wpos = (const float*)d_in[8];

  char* ws = (char*)d_ws;
  size_t off = 0;
  auto alloc = [&](size_t bytes) -> char* {
    char* p = ws + off;
    off += (bytes + 255) & ~(size_t)255;
    return p;
  };
  const size_t wmatTot = (size_t)HH * EE * DD * PP;          // 8.39M elems
  bf16*  xbf    = (bf16*)alloc((size_t)BT * DD * 2);
  bf16*  pebf   = (bf16*)alloc((size_t)1024 * DD * 2);        // padded rows
  bf16*  wposbf = (bf16*)alloc((size_t)1024 * DD * 2);
  bf16*  wqp    = (bf16*)alloc(wmatTot * 2);
  bf16*  wkp    = (bf16*)alloc(wmatTot * 2);
  bf16*  wvp    = (bf16*)alloc(wmatTot * 2);
  bf16*  wop    = (bf16*)alloc((size_t)HH * EE * PVV * DD * 2);
  float* go     = (float*)alloc((size_t)BT * HH * EE * 4);
  float* gv     = (float*)alloc((size_t)BT * HH * EE * 4);
  bf16*  qbf    = (bf16*)alloc((size_t)BT * HH * PP * 2);
  bf16*  kbf    = (bf16*)alloc((size_t)BT * HH * PP * 2);
  bf16*  vTbf   = (bf16*)alloc((size_t)BB * HH * PVV * TT * 2);
  bf16*  kposb  = (bf16*)alloc((size_t)1024 * HH * PP * 2);
  float* ctx    = (float*)alloc((size_t)BT * HH * PVV * 4);

  // converts (pe zero-padded to 1024 rows)
  cvt_kernel<<<(BT * DD + 255) / 256, 256, 0, stream>>>(x, xbf, BT * DD, BT * DD);
  cvt_kernel<<<(1024 * DD + 255) / 256, 256, 0, stream>>>(pe, pebf, RR * DD, 1024 * DD);
  cvt_kernel<<<(1024 * DD + 255) / 256, 256, 0, stream>>>(Wpos, wposbf, 1024 * DD,
                                                          1024 * DD);
  // weight packs: W_{q,k,v}[h,e] is (D,P) row-major -> sk=P, sn=1
  int totQKV = (int)wmatTot;
  pack_b_kernel<<<(totQKV + 255) / 256, 256, 0, stream>>>(Wq, wqp, PP, PP, 1,
                                                          DD * PP, totQKV);
  pack_b_kernel<<<(totQKV + 255) / 256, 256, 0, stream>>>(Wk, wkp, PP, PP, 1,
                                                          DD * PP, totQKV);
  pack_b_kernel<<<(totQKV + 255) / 256, 256, 0, stream>>>(Wv, wvp, PP, PP, 1,
                                                          DD * PP, totQKV);
  // W_o[h,e] is (PV,D) row-major -> K=PV, N=D, sk=D, sn=1
  pack_b_kernel<<<(totQKV + 255) / 256, 256, 0, stream>>>(Wo, wop, DD, DD, 1,
                                                          PVV * DD, totQKV);
  // gating (both gate sets)
  gating_kernel<<<(BT * HH) / 256, 256, 0, stream>>>(x, wso, wsv, go, gv);

  // q / k / v MoE projections
  dim3 gq(BT / 16, HH);
  moe_proj_kernel<<<gq, 32, 0, stream>>>(xbf, wqp, go, qbf, 0);
  moe_proj_kernel<<<gq, 32, 0, stream>>>(xbf, wkp, gv, kbf, 0);
  moe_proj_kernel<<<gq, 32, 0, stream>>>(xbf, wvp, gv, vTbf, 1);

  // positional keys
  kpos_kernel<<<dim3(64, 64), 32, 0, stream>>>(pebf, wposbf, kposb);

  // attention
  attn_kernel<<<dim3(TT / 16, HH, BB), 32, 0, stream>>>(qbf, kbf, vTbf, kposb, ctx);

  // output MoE projection
  out_kernel<<<dim3(BT / 16, DD / 64), 32, 0, stream>>>(ctx, go, wop, (float*)d_out);
}
